// DisentangledStateEncoder_86706799771777
// MI455X (gfx1250) — matrix operand order repaired and checked by reading, not compiled, not run on hardware
//
#include <hip/hip_runtime.h>
#include <hip/hip_bf16.h>

typedef __attribute__((ext_vector_type(16))) _Float16 v16h;
typedef __attribute__((ext_vector_type(8)))  _Float16 v8h;
typedef __attribute__((ext_vector_type(8)))  float    v8f;

#define B_DIM 512
#define T_DIM 512
#define RTOT  (B_DIM * T_DIM)   // 262144 rows
#define D_IN  128
#define H_DIM 64
#define S_DIM 5

__device__ __forceinline__ float gelu_exact(float v) {
    return 0.5f * v * (1.0f + erff(v * 0.70710678118654752440f));
}
__device__ __forceinline__ float sigmoidf_(float v) {
    return 1.0f / (1.0f + expf(-v));
}

// ---------------------------------------------------------------------------
// Phase 1: per row r of x (B*T x 128):
//   t   = x[r,:] @ W1 + b1                (WMMA f16, f32 acc)
//   h   = gelu(LN(t))                     (half-wave shuffle reduction)
//   pre[r,:]  = h @ Wc1[5:,:] + bc1       (WMMA via LDS re-layout of h)
//   ball[r,:] = h @ W_inn + b_inn         (WMMA, N padded 5->16)
// Block = 128 threads = 4 waves; each wave owns a 16-row tile, 4 tiles/wave.
// ---------------------------------------------------------------------------
__global__ __launch_bounds__(128) void phase1_kernel(
    const float* __restrict__ x,    const float* __restrict__ W1,
    const float* __restrict__ b1,   const float* __restrict__ ln_g,
    const float* __restrict__ ln_b, const float* __restrict__ W_inn,
    const float* __restrict__ b_inn,const float* __restrict__ Wc1,
    const float* __restrict__ bc1,
    float* __restrict__ pre_ws, float* __restrict__ ball_ws)
{
    // B-operand packed tiles: [tile][lane][16 halves], lane's 16 halves contiguous.
    __shared__ __align__(32) _Float16 sW1h[4 * 4 * 32 * 16];   // 16 KB: W1 (K=128,N=64)
    __shared__ __align__(32) _Float16 sWc1hh[2 * 4 * 32 * 16]; //  8 KB: Wc1_h (K=64,N=64)
    __shared__ __align__(32) _Float16 sWinnh[2 * 32 * 16];     //  2 KB: W_inn (K=64,N=16 pad)
    __shared__ __align__(32) _Float16 hstage[4][16][64];       //  8 KB: h tile per wave

    const int tid = threadIdx.x;

    // ---- pack weights into B-operand layout -------------------------------
    // For lane L in tile (kk,n): halves e=0..15 correspond to K = 32*kk + 8*(L/16) + (e&7) + 16*(e>>3),
    // column N = 16*n + (L%16).
    for (int idx = tid; idx < 4 * 4 * 32 * 16; idx += 128) {
        int e = idx & 15, lane = (idx >> 4) & 31, n = (idx >> 9) & 3, kk = idx >> 11;
        int k = 32 * kk + 8 * (lane >> 4) + (e & 7) + 16 * (e >> 3);
        int col = 16 * n + (lane & 15);
        sW1h[idx] = (_Float16)W1[k * H_DIM + col];
    }
    for (int idx = tid; idx < 2 * 4 * 32 * 16; idx += 128) {
        int e = idx & 15, lane = (idx >> 4) & 31, n = (idx >> 9) & 3, kk = idx >> 11;
        int k = 32 * kk + 8 * (lane >> 4) + (e & 7) + 16 * (e >> 3);
        int col = 16 * n + (lane & 15);
        sWc1hh[idx] = (_Float16)Wc1[(5 + k) * H_DIM + col];   // Wc1_h = Wc1[5:,:]
    }
    for (int idx = tid; idx < 2 * 32 * 16; idx += 128) {
        int e = idx & 15, lane = (idx >> 4) & 31, kk = idx >> 9;
        int k = 32 * kk + 8 * (lane >> 4) + (e & 7) + 16 * (e >> 3);
        int col = lane & 15;
        sWinnh[idx] = (col < S_DIM) ? (_Float16)W_inn[k * S_DIM + col] : (_Float16)0.0f;
    }
    __syncthreads();

    const int lane = tid & 31;
    const int w    = tid >> 5;
    const int half = lane >> 4;   // 0: lanes 0-15, 1: lanes 16-31
    const int lm   = lane & 15;

    float b1c[4], gC[4], bC[4], bc1c[4];
#pragma unroll
    for (int n = 0; n < 4; ++n) {
        int col = 16 * n + lm;
        b1c[n] = b1[col]; gC[n] = ln_g[col]; bC[n] = ln_b[col]; bc1c[n] = bc1[col];
    }
    float binnc = (lm < S_DIM) ? b_inn[lm] : 0.0f;

    for (int it = 0; it < 4; ++it) {
        const int rowBase = blockIdx.x * 256 + it * 64 + w * 16;
        const float* xrow = x + (size_t)(rowBase + lm) * D_IN;

        v8f acc[4];
#pragma unroll
        for (int n = 0; n < 4; ++n)
#pragma unroll
            for (int v = 0; v < 8; ++v) acc[n][v] = 0.0f;

        // ---- main GEMM: x @ W1, K=128 in 4 steps of 32 --------------------
#pragma unroll
        for (int kk = 0; kk < 4; ++kk) {
            // A layout (16-bit 16x32): lanes0-15 row=lm hold K {0..7,16..23},
            // lanes16-31 row=lm hold K {8..15,24..31}.
            const int c0 = 32 * kk + 8 * half;
            const int c1 = c0 + 16;
            const float* xp0 = xrow + c0;
            const float* xp1 = xrow + c1;
            v16h a;
#pragma unroll
            for (int i = 0; i < 8; ++i) {
                a[i]     = (_Float16)xp0[i];
                a[i + 8] = (_Float16)xp1[i];
            }
#pragma unroll
            for (int n = 0; n < 4; ++n) {
                const v16h b = *reinterpret_cast<const v16h*>(
                    &sW1h[((((kk << 2) | n) * 32) + lane) << 4]);
                acc[n] = __builtin_amdgcn_wmma_f32_16x16x32_f16(
                    false, a, false, b, (short)0, acc[n], false, false);
            }
        }

        // ---- bias + LayerNorm + GELU -> hstage (f16, row-major) -----------
        // C layout: element v, lane L -> row m = v + 8*half, col = 16*n + lm
#pragma unroll
        for (int v = 0; v < 8; ++v) {
            float s = 0.0f, q = 0.0f;
#pragma unroll
            for (int n = 0; n < 4; ++n) {
                float t0 = acc[n][v] + b1c[n];
                acc[n][v] = t0;
                s += t0; q += t0 * t0;
            }
#pragma unroll
            for (int off = 1; off < 16; off <<= 1) {
                s += __shfl_xor(s, off, 16);
                q += __shfl_xor(q, off, 16);
            }
            float mean = s * (1.0f / 64.0f);
            float var  = q * (1.0f / 64.0f) - mean * mean;
            float rstd = rsqrtf(var + 1e-5f);
#pragma unroll
            for (int n = 0; n < 4; ++n) {
                float y = (acc[n][v] - mean) * rstd * gC[n] + bC[n];
                hstage[w][v + 8 * half][16 * n + lm] = (_Float16)gelu_exact(y);
            }
        }
        __syncthreads();

        // ---- pre = h @ Wc1_h + bc1 ; ball = h @ W_inn + b_inn -------------
        v8f cp[4]; v8f cb;
#pragma unroll
        for (int n = 0; n < 4; ++n)
#pragma unroll
            for (int v = 0; v < 8; ++v) cp[n][v] = bc1c[n];
#pragma unroll
        for (int v = 0; v < 8; ++v) cb[v] = binnc;

#pragma unroll
        for (int kk2 = 0; kk2 < 2; ++kk2) {
            const int c0 = 32 * kk2 + 8 * half;
            const int c1 = c0 + 16;
            v8h lo = *reinterpret_cast<const v8h*>(&hstage[w][lm][c0]);
            v8h hi = *reinterpret_cast<const v8h*>(&hstage[w][lm][c1]);
            v16h a;
#pragma unroll
            for (int i = 0; i < 8; ++i) { a[i] = lo[i]; a[i + 8] = hi[i]; }
#pragma unroll
            for (int n = 0; n < 4; ++n) {
                const v16h b = *reinterpret_cast<const v16h*>(
                    &sWc1hh[((((kk2 << 2) | n) * 32) + lane) << 4]);
                cp[n] = __builtin_amdgcn_wmma_f32_16x16x32_f16(
                    false, a, false, b, (short)0, cp[n], false, false);
            }
            const v16h b2 = *reinterpret_cast<const v16h*>(
                &sWinnh[((kk2 * 32) + lane) << 4]);
            cb = __builtin_amdgcn_wmma_f32_16x16x32_f16(
                false, a, false, b2, (short)0, cb, false, false);
        }

#pragma unroll
        for (int n = 0; n < 4; ++n)
#pragma unroll
            for (int v = 0; v < 8; ++v)
                pre_ws[(size_t)(rowBase + v + 8 * half) * H_DIM + 16 * n + lm] = cp[n][v];
        if (lm < S_DIM) {
#pragma unroll
            for (int v = 0; v < 8; ++v)
                ball_ws[(size_t)(rowBase + v + 8 * half) * S_DIM + lm] = cb[v];
        }
        __syncthreads();
    }
}

// ---------------------------------------------------------------------------
// Phase 2: sequential scan over T=512. 32 blocks x 16 batch rows, 256 threads.
// ---------------------------------------------------------------------------
__global__ __launch_bounds__(256) void phase2_kernel(
    const float* __restrict__ ball, const float* __restrict__ pre,
    const float* __restrict__ Wc1,  const float* __restrict__ Wc2,
    const float* __restrict__ bc2,  const float* __restrict__ corr_scale,
    const float* __restrict__ raw_aL, const float* __restrict__ raw_aT,
    const float* __restrict__ raw_g,  const float* __restrict__ raw_aR,
    const float* __restrict__ omega,  float* __restrict__ out)
{
    __shared__ float sWc1s[5][64];   // Wc1[:5,:]
    __shared__ float sWc2[64][5];
    __shared__ float sbc2[5];
    __shared__ float sstate[16][5];
    __shared__ float slin[16][5];
    __shared__ float shm[16][65];    // padded to kill bank conflicts

    const int tid = threadIdx.x;
    for (int i = tid; i < 5 * 64; i += 256) sWc1s[i / 64][i % 64] = Wc1[i];
    for (int i = tid; i < 64 * 5; i += 256) sWc2[i / 5][i % 5] = Wc2[i];
    if (tid < 5)  sbc2[tid] = bc2[tid];
    if (tid < 80) sstate[tid / 5][tid % 5] = 0.0f;

    const float cs = corr_scale[0];
    const float aL = sigmoidf_(raw_aL[0]) * 0.15f + 0.85f;
    const float aT = sigmoidf_(raw_aT[0]) * 0.25f + 0.70f;
    const float gg = sigmoidf_(raw_g[0])  * 0.20f + 0.80f;
    const float aR = sigmoidf_(raw_aR[0]) * 0.40f;
    const float om = omega[0];
    const float gc = gg * cosf(om);
    const float gs = gg * sinf(om);

    const int b0 = blockIdx.x * 16;
    __syncthreads();

    // register prefetch of pre_t (each thread owns 4 (m,j) cells)
    float pv[4];
#pragma unroll
    for (int k = 0; k < 4; ++k) {
        int idx = k * 256 + tid, m = idx >> 6, j = idx & 63;
        pv[k] = pre[((size_t)(b0 + m) * T_DIM + 0) * H_DIM + j];
    }

    for (int t = 0; t < T_DIM; ++t) {
        // stage A: gates + s_lin
        if (tid < 80) {
            int m = tid / 5, i = tid % 5;
            float s0 = sstate[m][0], s1 = sstate[m][1], s2 = sstate[m][2];
            float s3 = sstate[m][3], s4 = sstate[m][4];
            float sA;
            if      (i == 0) sA = s0 * aL;
            else if (i == 1) sA = s1 * aT;
            else if (i == 2) sA = s2 * gc + s3 * gs;
            else if (i == 3) sA = -s2 * gs + s3 * gc;
            else             sA = s4 * aR;
            slin[m][i] = sA + ball[((size_t)(b0 + m) * T_DIM + t) * S_DIM + i];
        }
        __syncthreads();

        // stage B: hmlp = gelu(s_lin @ Wc1_s + pre_t)
#pragma unroll
        for (int k = 0; k < 4; ++k) {
            int idx = k * 256 + tid, m = idx >> 6, j = idx & 63;
            float v = pv[k];
#pragma unroll
            for (int kk = 0; kk < 5; ++kk) v += slin[m][kk] * sWc1s[kk][j];
            shm[m][j] = gelu_exact(v);
        }
        // prefetch next step's pre into registers (+deep L2 prefetch hint)
        if (t + 1 < T_DIM) {
#pragma unroll
            for (int k = 0; k < 4; ++k) {
                int idx = k * 256 + tid, m = idx >> 6, j = idx & 63;
                pv[k] = pre[((size_t)(b0 + m) * T_DIM + (t + 1)) * H_DIM + j];
            }
        }
        if (tid < 64 && t + 8 < T_DIM) {
            int m = tid >> 2, off = (tid & 3) * 16;
            __builtin_prefetch(&pre[((size_t)(b0 + m) * T_DIM + (t + 8)) * H_DIM + off], 0, 0);
        }
        __syncthreads();

        // stage C: s = s_lin + cs * tanh(hmlp @ Wc2 + bc2)
        if (tid < 80) {
            int m = tid / 5, i = tid % 5;
            float acc = sbc2[i];
#pragma unroll
            for (int j = 0; j < 64; ++j) acc += shm[m][j] * sWc2[j][i];
            sstate[m][i] = slin[m][i] + cs * tanhf(acc);
        }
        __syncthreads();
    }

    if (tid < 80) out[(b0 + tid / 5) * S_DIM + tid % 5] = sstate[tid / 5][tid % 5];
}

extern "C" void kernel_launch(void* const* d_in, const int* in_sizes, int n_in,
                              void* d_out, int out_size, void* d_ws, size_t ws_size,
                              hipStream_t stream) {
    const float* x     = (const float*)d_in[0];
    const float* W1    = (const float*)d_in[1];
    const float* b1    = (const float*)d_in[2];
    const float* ln_g  = (const float*)d_in[3];
    const float* ln_b  = (const float*)d_in[4];
    const float* W_inn = (const float*)d_in[5];
    const float* b_inn = (const float*)d_in[6];
    const float* Wc1   = (const float*)d_in[7];
    const float* bc1   = (const float*)d_in[8];
    const float* Wc2   = (const float*)d_in[9];
    const float* bc2   = (const float*)d_in[10];
    const float* corr_scale = (const float*)d_in[11];
    const float* raw_aL = (const float*)d_in[12];
    const float* raw_aT = (const float*)d_in[13];
    const float* raw_g  = (const float*)d_in[14];
    const float* raw_aR = (const float*)d_in[15];
    const float* omega  = (const float*)d_in[16];

    float* pre_ws  = (float*)d_ws;                        // RTOT*64 floats (64 MB)
    float* ball_ws = pre_ws + (size_t)RTOT * H_DIM;       // RTOT*5 floats  (5 MB)

    phase1_kernel<<<1024, 128, 0, stream>>>(x, W1, b1, ln_g, ln_b, W_inn, b_inn,
                                            Wc1, bc1, pre_ws, ball_ws);
    phase2_kernel<<<32, 256, 0, stream>>>(ball_ws, pre_ws, Wc1, Wc2, bc2,
                                          corr_scale, raw_aL, raw_aT, raw_g,
                                          raw_aR, omega, (float*)d_out);
}
